// Net_81939386073094
// MI455X (gfx1250) — matrix operand order repaired and checked by reading, not compiled, not run on hardware
//
#include <hip/hip_runtime.h>

// ============================================================================
// SINDy autoencoder on MI455X (gfx1250, wave32, WMMA f16 16x16x32).
//
//  1. H0 = sigmoid(x @ enc_w0 + b0)            WMMA (A f32->f16), store f16
//  2. H1 = sigmoid(H0 @ enc_w1 + b1)           WMMA (A f16, async->LDS)
//  3. z, theta, dzb, G0 = sigmoid(z@dec_w0+b)  VALU per-row
//  4. G1 = sigmoid(G0 @ dec_w1 + b1)           WMMA
//  5. xb = G1 @ dec_w2 + b2                    WMMA, f32 -> d_out
//  6. S-partials = H0'^T H1'  (sigma' = h(1-h) on staging)   WMMA
//  7. T-partials = G0'^T G1'                                 WMMA
//  8. reduce partials -> U = enc_w1 .* S/N,  V = dec_w1 .* T/N
//  9. Kd = V @ dec_w2 [128,512], Menc = U @ enc_w2 [256,3]
// 10. Je = (enc_w0 @ Menc)^T [3,512], JdT = dec_w0 @ Kd [3,512]
// 11. dz = dx @ Je^T, dxb = dzb @ JdT          VALU, bandwidth-bound
// ============================================================================

typedef __attribute__((ext_vector_type(16))) _Float16 v16h;
typedef __attribute__((ext_vector_type(8)))  _Float16 v8h;
typedef __attribute__((ext_vector_type(2)))  _Float16 v2h;
typedef __attribute__((ext_vector_type(8)))  float    v8f;

#define N_ROWS   65536
#define IN_DIM   512
#define H1_DIM   256
#define H2_DIM   128
#define NPART    64
#define ROWS_PER_PART (N_ROWS / NPART)

// Async global->LDS (ASYNCcnt path), gated so absence degrades gracefully.
#if defined(__has_builtin)
#if __has_builtin(__builtin_amdgcn_global_load_async_to_lds_b128)
#define HAVE_ASYNC_LDS 1
#endif
#endif
#ifndef HAVE_ASYNC_LDS
#define HAVE_ASYNC_LDS 0
#endif

#if HAVE_ASYNC_LDS
// Exact parameter types per the compiler diagnostic:
//   arg0: int __attribute__((vector_size(16))) addrspace(1)*  (printed "__device__")
//   arg1: LDS-side int4 pointer (addrspace(3))
typedef int i4vec __attribute__((vector_size(16)));
typedef __attribute__((address_space(1))) i4vec async_g_t;
typedef __attribute__((address_space(3))) i4vec async_l_t;
#endif

__device__ __forceinline__ float sigmoidf_(float x) { return 1.f / (1.f + __expf(-x)); }

constexpr int log2c(int x) { return x <= 1 ? 0 : 1 + log2c(x >> 1); }

// ---- WMMA fragment loaders (ISA 7.12.2 layouts, wave32) --------------------
// A (16x32 f16): lane holds row M=(lane&15); halves 0..7 = K base (lane>=16?8:0),
// halves 8..15 = 16 + same. Tile row-major, ld = 32 -> two ds_load_b128.
__device__ __forceinline__ v16h frag_a_ld(const _Float16* tile, int rowBase, int lane) {
    int r = rowBase + (lane & 15);
    int g = (lane >> 4) << 3;
    const v8h* p0 = (const v8h*)(tile + r * 32 + g);
    const v8h* p1 = (const v8h*)(tile + r * 32 + 16 + g);
    v8h lo = *p0, hi = *p1;
    v16h a;
#pragma unroll
    for (int i = 0; i < 8; ++i) { a[i] = lo[i]; a[i + 8] = hi[i]; }
    return a;
}
// B (32x16 f16): lane holds col N=(lane&15); halves 0..7 = K base (lane>=16?8:0),
// halves 8..15 = 16 + same.
// Tile stored transposed: tileT[col][32 K], ld = 32 -> two ds_load_b128.
__device__ __forceinline__ v16h frag_b_ld(const _Float16* tileT, int colBase, int lane) {
    int c = colBase + (lane & 15);
    int g = (lane >> 4) << 3;
    const v8h* p0 = (const v8h*)(tileT + c * 32 + g);
    const v8h* p1 = (const v8h*)(tileT + c * 32 + 16 + g);
    v8h lo = *p0, hi = *p1;
    v16h b;
#pragma unroll
    for (int i = 0; i < 8; ++i) { b[i] = lo[i]; b[i + 8] = hi[i]; }
    return b;
}

// ---- Tiled GEMM: C[M,N] = act(A[M,K] @ W[K,N] + bias), compile-time shapes -
// BLOCK_M = 128 rows (8 waves x 16-row M-tiles); NC = N-chunk (128 or 256).
// ACT==1: sigmoid -> f16 output.  ACT==0: f32 output (ld = N).
template <bool AF32, int ACT, int K, int N, int NC>
__global__ __launch_bounds__(256) void wmma_gemm(const float* __restrict__ A32,
                                                 const _Float16* __restrict__ A16,
                                                 const float* __restrict__ W,
                                                 const float* __restrict__ bias,
                                                 _Float16* __restrict__ Ch,
                                                 float* __restrict__ Cf) {
    constexpr int NT = NC / 16;            // 16x16 N-tiles per wave
    constexpr int LOG2NC = log2c(NC);
    __shared__ _Float16 As[128 * 32];
    __shared__ _Float16 Wt[NC * 32];
    const int tid = threadIdx.x, lane = tid & 31, wave = tid >> 5;
    const int mBase = blockIdx.x * 128;
    const int nBase = blockIdx.y * NC;

    v8f acc[NT];
#pragma unroll
    for (int t = 0; t < NT; ++t)
#pragma unroll
        for (int r = 0; r < 8; ++r) acc[t][r] = 0.f;

    const int rowS = tid >> 1;             // staging: 2 threads per row
    const int kkS  = (tid & 1) << 4;       // 0 or 16

    for (int k0 = 0; k0 < K; k0 += 32) {
        // ---- stage A tile [128 x 32] f16, row-major ld=32 ----
        if constexpr (AF32) {
            const float4* src = (const float4*)(A32 + (size_t)(mBase + rowS) * K + k0 + kkS);
            if (k0 + 32 < K)   // cover HBM latency on the streaming input
                __builtin_prefetch(A32 + (size_t)(mBase + rowS) * K + k0 + 32 + kkS, 0, 1);
            float4 f0 = src[0], f1 = src[1], f2 = src[2], f3 = src[3];
            v8h h0, h1;
            h0[0] = (_Float16)f0.x; h0[1] = (_Float16)f0.y; h0[2] = (_Float16)f0.z; h0[3] = (_Float16)f0.w;
            h0[4] = (_Float16)f1.x; h0[5] = (_Float16)f1.y; h0[6] = (_Float16)f1.z; h0[7] = (_Float16)f1.w;
            h1[0] = (_Float16)f2.x; h1[1] = (_Float16)f2.y; h1[2] = (_Float16)f2.z; h1[3] = (_Float16)f2.w;
            h1[4] = (_Float16)f3.x; h1[5] = (_Float16)f3.y; h1[6] = (_Float16)f3.z; h1[7] = (_Float16)f3.w;
            v8h* dst = (v8h*)(As + rowS * 32 + kkS);
            dst[0] = h0; dst[1] = h1;
        } else {
#if HAVE_ASYNC_LDS
            async_g_t* gp = (async_g_t*)(uintptr_t)(A16 + (size_t)(mBase + rowS) * K + k0 + kkS);
            async_l_t* lp = (async_l_t*)(unsigned)(uintptr_t)(As + rowS * 32 + kkS);
            __builtin_amdgcn_global_load_async_to_lds_b128(gp, lp, 0, 0);
            __builtin_amdgcn_global_load_async_to_lds_b128(gp + 1, lp + 1, 0, 0);
#else
            const v8h* src = (const v8h*)(A16 + (size_t)(mBase + rowS) * K + k0 + kkS);
            v8h* dst = (v8h*)(As + rowS * 32 + kkS);
            dst[0] = src[0]; dst[1] = src[1];
#endif
        }
        // ---- stage W chunk transposed: Wt[col][k], two K per thread -> b32 stores
#pragma unroll
        for (int it = 0; it < (16 * NC) / 256; ++it) {
            int idx = tid + it * 256;
            int kk  = (idx >> LOG2NC) << 1;
            int c   = idx & (NC - 1);
            v2h o;
            o[0] = (_Float16)W[(size_t)(k0 + kk + 0) * N + nBase + c];
            o[1] = (_Float16)W[(size_t)(k0 + kk + 1) * N + nBase + c];
            *(v2h*)(Wt + c * 32 + kk) = o;
        }
#if HAVE_ASYNC_LDS
        if constexpr (!AF32) asm volatile("s_wait_asynccnt 0" ::: "memory");
#endif
        __syncthreads();

        v16h a = frag_a_ld(As, wave * 16, lane);
#pragma unroll
        for (int t = 0; t < NT; ++t) {
            v16h b = frag_b_ld(Wt, t * 16, lane);
            acc[t] = __builtin_amdgcn_wmma_f32_16x16x32_f16(
                false, a, false, b, (short)0, acc[t], false, false);
        }
        __syncthreads();
    }

    // Writeback (C/D layout: vgpr r -> row r + 8*(lane>>4), col = lane&15).
#pragma unroll
    for (int t = 0; t < NT; ++t) {
        const int colB = nBase + t * 16 + (lane & 15);
        const int rowB = mBase + wave * 16 + ((lane >> 4) << 3);
        const float bv = bias[colB];
#pragma unroll
        for (int r = 0; r < 8; ++r) {
            float v = acc[t][r] + bv;
            const size_t o = (size_t)(rowB + r) * N + colB;
            if (ACT == 1) Ch[o] = (_Float16)sigmoidf_(v);
            else          Cf[o] = v;
        }
    }
}

// ---- A'^T B' GEMM for Jacobian second moments ------------------------------
// Partial C[P,Q] = sum_rows a'(A[n,i]) * b'(B[n,j]),  a'(h)=h*(1-h) on staging.
template <int TP, int TQ>
__global__ __launch_bounds__(256) void atb_kernel(const _Float16* __restrict__ Ag,
                                                  const _Float16* __restrict__ Bg,
                                                  float* __restrict__ Part) {
    constexpr int P = TP * 16, Q = TQ * 16;
    constexpr int LOG2P = log2c(P), LOG2Q = log2c(Q);
    __shared__ _Float16 At[P * 32];
    __shared__ _Float16 Bt[Q * 32];
    const int tid = threadIdx.x, lane = tid & 31, wave = tid >> 5;
    const size_t rowBase = (size_t)blockIdx.x * ROWS_PER_PART;

    v8f acc[16];
#pragma unroll
    for (int t = 0; t < 16; ++t)
#pragma unroll
        for (int r = 0; r < 8; ++r) acc[t][r] = 0.f;

    for (int k0 = 0; k0 < ROWS_PER_PART; k0 += 32) {
#pragma unroll
        for (int it = 0; it < (16 * P) / 256; ++it) {
            int idx = tid + it * 256;
            int n   = (idx >> LOG2P) << 1;
            int i   = idx & (P - 1);
            float ha = (float)Ag[(rowBase + k0 + n + 0) * P + i];
            float hb = (float)Ag[(rowBase + k0 + n + 1) * P + i];
            v2h o; o[0] = (_Float16)(ha * (1.f - ha)); o[1] = (_Float16)(hb * (1.f - hb));
            *(v2h*)(At + i * 32 + n) = o;
        }
#pragma unroll
        for (int it = 0; it < (16 * Q) / 256; ++it) {
            int idx = tid + it * 256;
            int n   = (idx >> LOG2Q) << 1;
            int j   = idx & (Q - 1);
            float ha = (float)Bg[(rowBase + k0 + n + 0) * Q + j];
            float hb = (float)Bg[(rowBase + k0 + n + 1) * Q + j];
            v2h o; o[0] = (_Float16)(ha * (1.f - ha)); o[1] = (_Float16)(hb * (1.f - hb));
            *(v2h*)(Bt + j * 32 + n) = o;
        }
        __syncthreads();
#pragma unroll
        for (int u = 0; u < 16; ++u) {
            int tile = wave * 16 + u;
            int tp = tile / TQ, tq = tile - tp * TQ;
            v16h a = frag_a_ld(At, tp * 16, lane);
            v16h b = frag_b_ld(Bt, tq * 16, lane);
            acc[u] = __builtin_amdgcn_wmma_f32_16x16x32_f16(
                false, a, false, b, (short)0, acc[u], false, false);
        }
        __syncthreads();
    }

    float* outp = Part + (size_t)blockIdx.x * (P * Q);
#pragma unroll
    for (int u = 0; u < 16; ++u) {
        int tile = wave * 16 + u;
        int tp = tile / TQ, tq = tile - tp * TQ;
        int gi = tp * 16 + ((lane >> 4) << 3);
        int gj = tq * 16 + (lane & 15);
#pragma unroll
        for (int r = 0; r < 8; ++r) outp[(gi + r) * Q + gj] = acc[u][r];
    }
}

// ---- per-row latent work: z, theta, dzb, G0 --------------------------------
__global__ __launch_bounds__(256) void mid_kernel(const _Float16* __restrict__ H1h,
                                                  const float* __restrict__ w2e, const float* __restrict__ b2e,
                                                  const float* __restrict__ Ew,  const float* __restrict__ Eb,
                                                  const float* __restrict__ w0d, const float* __restrict__ b0d,
                                                  float* __restrict__ zO, float* __restrict__ dzbO,
                                                  _Float16* __restrict__ G0h) {
    const int n = blockIdx.x * 256 + threadIdx.x;
    const _Float16* hp = H1h + (size_t)n * H2_DIM;
    float z0 = b2e[0], z1 = b2e[1], z2 = b2e[2];
    for (int k = 0; k < H2_DIM; ++k) {
        float h = (float)hp[k];
        z0 += h * w2e[k * 3 + 0]; z1 += h * w2e[k * 3 + 1]; z2 += h * w2e[k * 3 + 2];
    }
    zO[n * 3 + 0] = z0; zO[n * 3 + 1] = z1; zO[n * 3 + 2] = z2;

    // sindy_library: ones(3), linear(3), quad i<=j (6), cubic i<=j<=k (10)
    float th[22] = {1.f, 1.f, 1.f, z0, z1, z2,
                    z0 * z0, z0 * z1, z0 * z2, z1 * z1, z1 * z2, z2 * z2,
                    z0 * z0 * z0, z0 * z0 * z1, z0 * z0 * z2, z0 * z1 * z1, z0 * z1 * z2,
                    z0 * z2 * z2, z1 * z1 * z1, z1 * z1 * z2, z1 * z2 * z2, z2 * z2 * z2};
#pragma unroll
    for (int l = 0; l < 3; ++l) {
        float s = Eb[l];
#pragma unroll
        for (int c = 0; c < 22; ++c) s += th[c] * Ew[c * 3 + l];
        dzbO[n * 3 + l] = s;
    }
    for (int j = 0; j < H2_DIM; ++j) {
        float v = b0d[j] + z0 * w0d[j] + z1 * w0d[H2_DIM + j] + z2 * w0d[2 * H2_DIM + j];
        G0h[(size_t)n * H2_DIM + j] = (_Float16)sigmoidf_(v);
    }
}

// ---- reduce partials, apply hadamard weights: U, V -------------------------
__global__ __launch_bounds__(256) void reduce_uv(const float* __restrict__ Sp, const float* __restrict__ Tp,
                                                 const float* __restrict__ w1e, const float* __restrict__ w1d,
                                                 float* __restrict__ U, float* __restrict__ V) {
    const int e = blockIdx.x * 256 + threadIdx.x;  // 0..65535
    const float invN = 1.f / (float)N_ROWS;
    if (e < H1_DIM * H2_DIM) {
        float s = 0.f;
        for (int p = 0; p < NPART; ++p) s += Sp[(size_t)p * (H1_DIM * H2_DIM) + e];
        U[e] = w1e[e] * s * invN;
    } else {
        const int e2 = e - H1_DIM * H2_DIM;
        float s = 0.f;
        for (int p = 0; p < NPART; ++p) s += Tp[(size_t)p * (H2_DIM * H1_DIM) + e2];
        V[e2] = w1d[e2] * s * invN;
    }
}

// ---- Kd = V @ dec_w2 [128,512];  Menc = U @ enc_w2 [256,3] -----------------
__global__ __launch_bounds__(256) void kd_menc(const float* __restrict__ V, const float* __restrict__ w2d,
                                               const float* __restrict__ U, const float* __restrict__ w2e,
                                               float* __restrict__ Kd, float* __restrict__ Menc) {
    const int blk = blockIdx.x, t = threadIdx.x;
    if (blk < 256) {
        const int idx = blk * 256 + t;
        const int j = idx >> 9, p = idx & 511;
        float s = 0.f;
        for (int i = 0; i < H1_DIM; ++i) s += V[j * H1_DIM + i] * w2d[(size_t)i * IN_DIM + p];
        Kd[idx] = s;
    } else {
        const int idx = (blk - 256) * 256 + t;
        if (idx < H1_DIM * 3) {
            const int i = idx / 3, l = idx - 3 * i;
            float s = 0.f;
            for (int j = 0; j < H2_DIM; ++j) s += U[i * H2_DIM + j] * w2e[j * 3 + l];
            Menc[idx] = s;
        }
    }
}

// ---- Je[3,512] and JdT[3,512] ----------------------------------------------
__global__ __launch_bounds__(256) void jac_kernel(const float* __restrict__ w0e, const float* __restrict__ Menc,
                                                  const float* __restrict__ w0d, const float* __restrict__ Kd,
                                                  float* __restrict__ Je, float* __restrict__ JdT) {
    const int blk = blockIdx.x, t = threadIdx.x;
    if (blk < 6) {
        const int idx = blk * 256 + t;
        const int l = idx >> 9, q = idx & 511;
        float s = 0.f;
        for (int i = 0; i < H1_DIM; ++i) s += w0e[(size_t)q * H1_DIM + i] * Menc[i * 3 + l];
        Je[idx] = s;
    } else {
        const int idx = (blk - 6) * 256 + t;
        const int l = idx >> 9, p = idx & 511;
        float s = 0.f;
        for (int j = 0; j < H2_DIM; ++j) s += w0d[l * H2_DIM + j] * Kd[(size_t)j * IN_DIM + p];
        JdT[idx] = s;
    }
}

// ---- dz = dx @ Je^T,  dxb = dzb @ JdT : one wave per row -------------------
__global__ __launch_bounds__(256) void dz_dxb_kernel(const float* __restrict__ dx,
                                                     const float* __restrict__ dzb,
                                                     const float* __restrict__ Je,
                                                     const float* __restrict__ JdT,
                                                     float* __restrict__ dz,
                                                     float* __restrict__ dxb) {
    __shared__ float sJe[3 * IN_DIM];
    __shared__ float sJd[3 * IN_DIM];
    for (int i = threadIdx.x; i < 3 * IN_DIM; i += 256) { sJe[i] = Je[i]; sJd[i] = JdT[i]; }
    __syncthreads();
    const int lane = threadIdx.x & 31, wave = threadIdx.x >> 5;
    const int n = blockIdx.x * 8 + wave;
    const float4* dxr = (const float4*)(dx + (size_t)n * IN_DIM);
    float a0 = 0.f, a1 = 0.f, a2 = 0.f;
    const int c0 = lane * 16;
#pragma unroll
    for (int v = 0; v < 4; ++v) {
        float4 d = dxr[lane * 4 + v];
        int c = c0 + v * 4;
        a0 += d.x * sJe[c] + d.y * sJe[c + 1] + d.z * sJe[c + 2] + d.w * sJe[c + 3];
        a1 += d.x * sJe[IN_DIM + c] + d.y * sJe[IN_DIM + c + 1] + d.z * sJe[IN_DIM + c + 2] + d.w * sJe[IN_DIM + c + 3];
        a2 += d.x * sJe[2 * IN_DIM + c] + d.y * sJe[2 * IN_DIM + c + 1] + d.z * sJe[2 * IN_DIM + c + 2] + d.w * sJe[2 * IN_DIM + c + 3];
    }
#pragma unroll
    for (int o = 16; o > 0; o >>= 1) {
        a0 += __shfl_down(a0, o, 32);
        a1 += __shfl_down(a1, o, 32);
        a2 += __shfl_down(a2, o, 32);
    }
    if (lane == 0) { dz[n * 3 + 0] = a0; dz[n * 3 + 1] = a1; dz[n * 3 + 2] = a2; }
    const float b0 = dzb[n * 3 + 0], b1 = dzb[n * 3 + 1], b2 = dzb[n * 3 + 2];
    float4* ox = (float4*)(dxb + (size_t)n * IN_DIM);
#pragma unroll
    for (int v = 0; v < 4; ++v) {
        int c = c0 + v * 4;
        float4 o;
        o.x = b0 * sJd[c + 0] + b1 * sJd[IN_DIM + c + 0] + b2 * sJd[2 * IN_DIM + c + 0];
        o.y = b0 * sJd[c + 1] + b1 * sJd[IN_DIM + c + 1] + b2 * sJd[2 * IN_DIM + c + 1];
        o.z = b0 * sJd[c + 2] + b1 * sJd[IN_DIM + c + 2] + b2 * sJd[2 * IN_DIM + c + 2];
        o.w = b0 * sJd[c + 3] + b1 * sJd[IN_DIM + c + 3] + b2 * sJd[2 * IN_DIM + c + 3];
        ox[lane * 4 + v] = o;
    }
}

// ============================================================================
extern "C" void kernel_launch(void* const* d_in, const int* in_sizes, int n_in,
                              void* d_out, int out_size, void* d_ws, size_t ws_size,
                              hipStream_t stream) {
    const float* x      = (const float*)d_in[0];
    const float* dx     = (const float*)d_in[1];
    // d_in[2] = ddx (unused by the reference)
    const float* enc_w0 = (const float*)d_in[3];
    const float* enc_b0 = (const float*)d_in[4];
    const float* enc_w1 = (const float*)d_in[5];
    const float* enc_b1 = (const float*)d_in[6];
    const float* enc_w2 = (const float*)d_in[7];
    const float* enc_b2 = (const float*)d_in[8];
    const float* dec_w0 = (const float*)d_in[9];
    const float* dec_b0 = (const float*)d_in[10];
    const float* dec_w1 = (const float*)d_in[11];
    const float* dec_b1 = (const float*)d_in[12];
    const float* dec_w2 = (const float*)d_in[13];
    const float* dec_b2 = (const float*)d_in[14];
    const float* E_w    = (const float*)d_in[15];
    const float* E_b    = (const float*)d_in[16];
    (void)in_sizes; (void)n_in; (void)out_size; (void)ws_size;

    float* out  = (float*)d_out;   // z | dz | dzb | xb | dxb (flat, row-major each)
    float* zO   = out;
    float* dzO  = out + (size_t)N_ROWS * 3;
    float* dzbO = out + (size_t)N_ROWS * 6;
    float* xbO  = out + (size_t)N_ROWS * 9;
    float* dxbO = xbO + (size_t)N_ROWS * IN_DIM;

    // Workspace carve-up (~118 MB total)
    char* ws = (char*)d_ws;
    size_t off = 0;
    auto take = [&](size_t bytes) -> char* {
        char* p = ws + off;
        off += (bytes + 255) & ~(size_t)255;
        return p;
    };
    _Float16* H0h = (_Float16*)take((size_t)N_ROWS * H1_DIM * 2);
    _Float16* H1h = (_Float16*)take((size_t)N_ROWS * H2_DIM * 2);
    _Float16* G0h = (_Float16*)take((size_t)N_ROWS * H2_DIM * 2);
    _Float16* G1h = (_Float16*)take((size_t)N_ROWS * H1_DIM * 2);
    float* Spart = (float*)take((size_t)NPART * H1_DIM * H2_DIM * 4);
    float* Tpart = (float*)take((size_t)NPART * H2_DIM * H1_DIM * 4);
    float* U     = (float*)take((size_t)H1_DIM * H2_DIM * 4);
    float* V     = (float*)take((size_t)H2_DIM * H1_DIM * 4);
    float* Menc  = (float*)take((size_t)H1_DIM * 3 * 4);
    float* Kd    = (float*)take((size_t)H2_DIM * IN_DIM * 4);
    float* Je    = (float*)take((size_t)3 * IN_DIM * 4);
    float* JdT   = (float*)take((size_t)3 * IN_DIM * 4);

    // 1. H0 = sigmoid(x @ enc_w0 + b0)   [65536,512]x[512,256]
    wmma_gemm<true, 1, IN_DIM, H1_DIM, 256><<<dim3(N_ROWS / 128, 1), 256, 0, stream>>>(
        x, nullptr, enc_w0, enc_b0, H0h, nullptr);
    // 2. H1 = sigmoid(H0 @ enc_w1 + b1)  [65536,256]x[256,128]
    wmma_gemm<false, 1, H1_DIM, H2_DIM, 128><<<dim3(N_ROWS / 128, 1), 256, 0, stream>>>(
        nullptr, H0h, enc_w1, enc_b1, H1h, nullptr);
    // 3. z, theta, dzb, G0
    mid_kernel<<<N_ROWS / 256, 256, 0, stream>>>(
        H1h, enc_w2, enc_b2, E_w, E_b, dec_w0, dec_b0, zO, dzbO, G0h);
    // 4. G1 = sigmoid(G0 @ dec_w1 + b1)  [65536,128]x[128,256]
    wmma_gemm<false, 1, H2_DIM, H1_DIM, 256><<<dim3(N_ROWS / 128, 1), 256, 0, stream>>>(
        nullptr, G0h, dec_w1, dec_b1, G1h, nullptr);
    // 5. xb = G1 @ dec_w2 + b2           [65536,256]x[256,512] -> f32 out
    wmma_gemm<false, 0, H1_DIM, IN_DIM, 256><<<dim3(N_ROWS / 128, 2), 256, 0, stream>>>(
        nullptr, G1h, dec_w2, dec_b2, nullptr, xbO);
    // 6. S partials = H0'^T H1'  [256,128]
    atb_kernel<16, 8><<<NPART, 256, 0, stream>>>(H0h, H1h, Spart);
    // 7. T partials = G0'^T G1'  [128,256]
    atb_kernel<8, 16><<<NPART, 256, 0, stream>>>(G0h, G1h, Tpart);
    // 8. U, V
    reduce_uv<<<256, 256, 0, stream>>>(Spart, Tpart, enc_w1, dec_w1, U, V);
    // 9. Kd, Menc
    kd_menc<<<259, 256, 0, stream>>>(V, dec_w2, U, enc_w2, Kd, Menc);
    // 10. Je, JdT
    jac_kernel<<<12, 256, 0, stream>>>(enc_w0, Menc, dec_w0, Kd, Je, JdT);
    // 11. dz, dxb
    dz_dxb_kernel<<<N_ROWS / 8, 256, 0, stream>>>(dx, dzbO, Je, JdT, dzO, dxbO);
}
